// SDFField_87385404604890
// MI455X (gfx1250) — matrix operand order repaired and checked by reading, not compiled, not run on hardware
//
#include <hip/hip_runtime.h>

// ---------------- constants ----------------
#define NPTS (2048 * 128)      // P = 262144
#define HIDD 256
#define GEO_ 15
#define GDIM 96
#define GD2  (96 * 96)
#define GD3  (96 * 96 * 96)

// workspace layout (bf16 element offsets)
constexpr int OFF_WT0  = 0;        // [256][64]  = W0^T padded (in 35->64)
constexpr int OFF_WT1  = 16384;    // [256][256] = W1^T
constexpr int OFF_WT2  = 81920;    // [16][256]  = W2^T
constexpr int OFF_WB1  = 86016;    // [256][256] = W1 (for dh2 @ W1^T)
constexpr int OFF_WB0  = 151552;   // [48][256]  = W0 padded (for dh1 @ W0^T)
constexpr int OFF_WTR0 = 163840;   // [256][64]  = Wr0^T padded (in 56->64)
constexpr int OFF_WTR1 = 180224;   // [16][256]  = Wr1^T padded (out 3->16)
constexpr int WS_TOTAL = 184320;   // bf16 elements (~360 KB)

// ---------------- types / helpers ----------------
typedef __attribute__((ext_vector_type(16))) __bf16        v16bf;
typedef __attribute__((ext_vector_type(8)))  float         v8f;
typedef __attribute__((ext_vector_type(8)))  unsigned int  v8u;

__device__ __forceinline__ float bf2f(unsigned short h) {
  return __uint_as_float(((unsigned)h) << 16);
}
__device__ __forceinline__ unsigned short f2bf(float f) {
  unsigned u = __float_as_uint(f);
  u += 0x7FFFu + ((u >> 16) & 1u);   // round-to-nearest-even
  return (unsigned short)(u >> 16);
}

// pack two 16-byte chunks of bf16 into a WMMA fragment register block
__device__ __forceinline__ v16bf mk_frag(const unsigned short* q0,
                                         const unsigned short* q1) {
  uint4 a = *reinterpret_cast<const uint4*>(q0);
  uint4 b = *reinterpret_cast<const uint4*>(q1);
  v8u u;
  u[0] = a.x; u[1] = a.y; u[2] = a.z; u[3] = a.w;
  u[4] = b.x; u[5] = b.y; u[6] = b.z; u[7] = b.w;
  return __builtin_bit_cast(v16bf, u);
}

// A fragment (16xK=32 bf16) from row-major [16][stride] buffer, col base kc.
// ISA 7.12.2 16-bit A layout: lanes 0-15 -> M, lanes 16-31 same M with K+8;
// VGPR0-3 = K 0..7(+8), VGPR4-7 = K 16..23(+8).
__device__ __forceinline__ v16bf load_a(const unsigned short* buf, int stride,
                                        int kc, int lane) {
  const int m  = lane & 15;
  const int kh = (lane >> 4) << 3;                  // 0 or 8
  const unsigned short* r = buf + m * stride + kc + kh;
  return mk_frag(r, r + 16);
}

// B fragment (K=32 x 16 bf16) from (N x K) row-major weight buffer,
// N-row base nb, K col base kc. Same K layout as A: lanes 16-31 use K+8; VGPR4-7 = K 16..23(+8).
__device__ __forceinline__ v16bf load_b(const unsigned short* w, int stride,
                                        int nb, int kc, int lane) {
  const int n  = lane & 15;
  const int kb = kc + ((lane >> 4) << 3);           // 0 or 8
  const unsigned short* r = w + (nb + n) * stride + kb;
  return mk_frag(r, r + 16);
}

__device__ __forceinline__ v8f wmma_bf16(v16bf a, v16bf b, v8f c) {
  return __builtin_amdgcn_wmma_f32_16x16x32_bf16(false, a, false, b,
                                                 (short)0, c, false, false);
}

// C layout: lanes 0-15 -> N=lane, M=r; lanes 16-31 -> N=lane-16, M=r+8
__device__ __forceinline__ void store_tile(unsigned short* buf, int stride,
                                           int nc, v8f acc, const float* bias,
                                           bool relu, int lane) {
  const int n  = lane & 15;
  const int mo = (lane >> 4) * 8;
  const float bb = bias ? bias[nc + n] : 0.f;
#pragma unroll
  for (int r = 0; r < 8; ++r) {
    float v = acc[r] + bb;
    if (relu) v = fmaxf(v, 0.f);
    buf[(mo + r) * stride + nc + n] = f2bf(v);
  }
}

// ---------------- weight prep: f32 -> bf16, transpose + pad ----------------
__global__ __launch_bounds__(256) void prep_weights(
    const float* __restrict__ W0, const float* __restrict__ W1,
    const float* __restrict__ W2, const float* __restrict__ Wr0,
    const float* __restrict__ Wr1, unsigned short* __restrict__ ws) {
  const int idx = blockIdx.x * blockDim.x + threadIdx.x;
  if (idx >= WS_TOTAL) return;
  float v = 0.f;
  if (idx < OFF_WT1) {                       // Wt0 [256][64] = W0^T
    int t = idx - OFF_WT0; int o = t >> 6, i = t & 63;
    v = (i < 35) ? W0[i * 256 + o] : 0.f;
  } else if (idx < OFF_WT2) {                // Wt1 [256][256] = W1^T
    int t = idx - OFF_WT1; int o = t >> 8, i = t & 255;
    v = W1[i * 256 + o];
  } else if (idx < OFF_WB1) {                // Wt2 [16][256] = W2^T
    int t = idx - OFF_WT2; int o = t >> 8, i = t & 255;
    v = W2[i * 16 + o];
  } else if (idx < OFF_WB0) {                // Wb1 [256][256] = W1
    int t = idx - OFF_WB1;
    v = W1[t];
  } else if (idx < OFF_WTR0) {               // Wb0 [48][256] = W0 padded
    int t = idx - OFF_WB0; int i = t >> 8, j = t & 255;
    v = (i < 35) ? W0[i * 256 + j] : 0.f;
  } else if (idx < OFF_WTR1) {               // Wtr0 [256][64] = Wr0^T
    int t = idx - OFF_WTR0; int o = t >> 6, i = t & 63;
    v = (i < 56) ? Wr0[i * 256 + o] : 0.f;
  } else {                                   // Wtr1 [16][256] = Wr1^T padded
    int t = idx - OFF_WTR1; int o = t >> 8, i = t & 255;
    v = (o < 3) ? Wr1[i * 3 + o] : 0.f;
  }
  ws[idx] = f2bf(v);
}

// ---------------- fused SDF field kernel: 1 wave32 per 16 points ----------
__global__ __launch_bounds__(32) void sdf_field_kernel(
    const float* __restrict__ points, const float* __restrict__ directions,
    const float* __restrict__ deltas, const float* __restrict__ vol0,
    const float* __restrict__ vol1, const float* __restrict__ b0,
    const float* __restrict__ b1, const float* __restrict__ b2,
    const float* __restrict__ br0, const float* __restrict__ br1,
    const float* __restrict__ W2, const float* __restrict__ beta,
    const float* __restrict__ variance,
    const unsigned short* __restrict__ ws, float* __restrict__ out) {
  const int lane = threadIdx.x;
  const int p0   = blockIdx.x * 16;
  constexpr int P = NPTS;

  __shared__ alignas(16) unsigned short Xb[16 * 64];     // MLP input (bf16)
  __shared__ alignas(16) unsigned short actA[16 * 256];
  __shared__ alignas(16) unsigned short actB[16 * 256];
  __shared__ alignas(16) unsigned short actC[16 * 256];
  __shared__ float featsS[16 * 32];
  __shared__ float Jx[16 * 32], Jy[16 * 32], Jz[16 * 32];
  __shared__ float outt[16 * 16];                        // sdf+geo head
  __shared__ float dxb[16 * 48];                         // dL/dx tiles
  __shared__ float gradS[16 * 3];

  // warm L2/L0 for the weight region (global_prefetch_b8)
  __builtin_prefetch(ws + (lane << 12), 0, 1);

  // ---- phase 1: trilinear gather + Jacobian (lane = point x volume) ----
  {
    const int pt = lane & 15;
    const int vs = lane >> 4;
    const float* __restrict__ vol = vs ? vol1 : vol0;
    const float px = points[(p0 + pt) * 3 + 0];
    const float py = points[(p0 + pt) * 3 + 1];
    const float pz = points[(p0 + pt) * 3 + 2];
    const float x = px * (float)(GDIM - 1);
    const float y = py * (float)(GDIM - 1);
    const float z = pz * (float)(GDIM - 1);
    const float xf = floorf(x), yf = floorf(y), zf = floorf(z);
    const int x0 = (int)xf, y0i = (int)yf, z0i = (int)zf;
    const float xd = x - xf, yd = y - yf, zd = z - zf;
    float f[16], gx[16], gy[16], gz[16];
#pragma unroll
    for (int c = 0; c < 16; ++c) { f[c] = 0.f; gx[c] = 0.f; gy[c] = 0.f; gz[c] = 0.f; }
#pragma unroll
    for (int corner = 0; corner < 8; ++corner) {
      const int dx = corner & 1, dy = (corner >> 1) & 1, dz = (corner >> 2) & 1;
      const int xi = x0 + dx, yi = y0i + dy, zi = z0i + dz;
      const bool valid = (xi >= 0) & (xi < GDIM) & (yi >= 0) & (yi < GDIM) &
                         (zi >= 0) & (zi < GDIM);
      const int xc = min(max(xi, 0), GDIM - 1);
      const int yc = min(max(yi, 0), GDIM - 1);
      const int zc = min(max(zi, 0), GDIM - 1);
      const float wx = dx ? xd : 1.f - xd;
      const float wy = dy ? yd : 1.f - yd;
      const float wz = dz ? zd : 1.f - zd;
      const float sx = dx ? 1.f : -1.f;
      const float sy = dy ? 1.f : -1.f;
      const float sz = dz ? 1.f : -1.f;
      const float w   = wx * wy * wz;
      const float dwx = sx * wy * wz;
      const float dwy = wx * sy * wz;
      const float dwz = wx * wy * sz;
      const float vm  = valid ? 1.f : 0.f;
      const float* base = vol + (size_t)zc * GD2 + (size_t)yc * GDIM + xc;
#pragma unroll
      for (int c = 0; c < 16; ++c) {
        const float v = base[(size_t)c * GD3] * vm;
        f[c] += w * v; gx[c] += dwx * v; gy[c] += dwy * v; gz[c] += dwz * v;
      }
    }
    const float sc = (float)(GDIM - 1);   // d(grid coord)/d(point)
#pragma unroll
    for (int c = 0; c < 16; ++c) {
      const int oi = (c < 8) ? (vs * 8 + c) : (16 + vs * 8 + (c - 8));
      featsS[pt * 32 + oi] = f[c];
      Jx[pt * 32 + oi] = gx[c] * sc;
      Jy[pt * 32 + oi] = gy[c] * sc;
      Jz[pt * 32 + oi] = gz[c] * sc;
    }
  }
  __syncthreads();

  // ---- phase 2: build SDF-MLP input X = [p(3), feats(32), pad] ----
  for (int i = lane; i < 16 * 64; i += 32) {
    const int pt = i >> 6, c = i & 63;
    float v = 0.f;
    if (c < 3)       v = points[(p0 + pt) * 3 + c];
    else if (c < 35) v = featsS[pt * 32 + (c - 3)];
    Xb[i] = f2bf(v);
  }
  __syncthreads();

  // ---- layer 1: X(16x64) @ W0 -> h1(16x256), relu ----
  {
    const v16bf a0 = load_a(Xb, 64, 0, lane);
    const v16bf a1 = load_a(Xb, 64, 32, lane);
#pragma unroll
    for (int nt = 0; nt < 16; ++nt) {
      v8f acc = {0.f, 0.f, 0.f, 0.f, 0.f, 0.f, 0.f, 0.f};
      acc = wmma_bf16(a0, load_b(ws + OFF_WT0, 64, nt * 16, 0, lane), acc);
      acc = wmma_bf16(a1, load_b(ws + OFF_WT0, 64, nt * 16, 32, lane), acc);
      store_tile(actA, 256, nt * 16, acc, b0, true, lane);
    }
  }
  __syncthreads();

  // ---- layer 2: h1 @ W1 -> h2(16x256), relu ----
  {
    v16bf h[8];
#pragma unroll
    for (int k = 0; k < 8; ++k) h[k] = load_a(actA, 256, k * 32, lane);
#pragma unroll
    for (int nt = 0; nt < 16; ++nt) {
      v8f acc = {0.f, 0.f, 0.f, 0.f, 0.f, 0.f, 0.f, 0.f};
#pragma unroll
      for (int k = 0; k < 8; ++k)
        acc = wmma_bf16(h[k], load_b(ws + OFF_WT1, 256, nt * 16, k * 32, lane), acc);
      store_tile(actB, 256, nt * 16, acc, b1, true, lane);
    }
  }
  __syncthreads();

  // ---- layer 3: h2 @ W2 + b2 -> outt(16x16)  (sdf | geo) ----
  {
    v16bf h[8];
#pragma unroll
    for (int k = 0; k < 8; ++k) h[k] = load_a(actB, 256, k * 32, lane);
    v8f acc = {0.f, 0.f, 0.f, 0.f, 0.f, 0.f, 0.f, 0.f};
#pragma unroll
    for (int k = 0; k < 8; ++k)
      acc = wmma_bf16(h[k], load_b(ws + OFF_WT2, 256, 0, k * 32, lane), acc);
    const int n = lane & 15, mo = (lane >> 4) * 8;
    const float bb = b2[n];
#pragma unroll
    for (int r = 0; r < 8; ++r) outt[(mo + r) * 16 + n] = acc[r] + bb;
  }
  __syncthreads();

  // ---- backward: dh2 = relu'(h2) * W2[:,0] ----
  for (int i = lane; i < 16 * 256; i += 32) {
    const int j = i & 255;
    const float h2v = bf2f(actB[i]);
    actC[i] = f2bf(h2v > 0.f ? W2[j * 16] : 0.f);
  }
  __syncthreads();

  // ---- dh1 = relu'(h1) * (dh2 @ W1^T) ----
  {
    v16bf h[8];
#pragma unroll
    for (int k = 0; k < 8; ++k) h[k] = load_a(actC, 256, k * 32, lane);
#pragma unroll
    for (int nt = 0; nt < 16; ++nt) {
      v8f acc = {0.f, 0.f, 0.f, 0.f, 0.f, 0.f, 0.f, 0.f};
#pragma unroll
      for (int k = 0; k < 8; ++k)
        acc = wmma_bf16(h[k], load_b(ws + OFF_WB1, 256, nt * 16, k * 32, lane), acc);
      const int n = lane & 15, mo = (lane >> 4) * 8;
#pragma unroll
      for (int r = 0; r < 8; ++r) {
        const int o = (mo + r) * 256 + nt * 16 + n;
        const float h1v = bf2f(actA[o]);
        actB[o] = f2bf(h1v > 0.f ? acc[r] : 0.f);
      }
    }
  }
  __syncthreads();

  // ---- dx = dh1 @ W0^T  (first 48 cols; 0..2 = dp, 3..34 = dfeat) ----
  {
    v16bf h[8];
#pragma unroll
    for (int k = 0; k < 8; ++k) h[k] = load_a(actB, 256, k * 32, lane);
#pragma unroll
    for (int nt = 0; nt < 3; ++nt) {
      v8f acc = {0.f, 0.f, 0.f, 0.f, 0.f, 0.f, 0.f, 0.f};
#pragma unroll
      for (int k = 0; k < 8; ++k)
        acc = wmma_bf16(h[k], load_b(ws + OFF_WB0, 256, nt * 16, k * 32, lane), acc);
      const int n = lane & 15, mo = (lane >> 4) * 8;
#pragma unroll
      for (int r = 0; r < 8; ++r) dxb[(mo + r) * 48 + nt * 16 + n] = acc[r];
    }
  }
  __syncthreads();

  // ---- per-point: gradients, normal, density, alpha, sdf ----
  if (lane < 16) {
    const int pt = lane, pi = p0 + pt;
    float g0 = dxb[pt * 48 + 0], g1 = dxb[pt * 48 + 1], g2 = dxb[pt * 48 + 2];
#pragma unroll
    for (int ff = 0; ff < 32; ++ff) {
      const float d = dxb[pt * 48 + 3 + ff];
      g0 += d * Jx[pt * 32 + ff];
      g1 += d * Jy[pt * 32 + ff];
      g2 += d * Jz[pt * 32 + ff];
    }
    gradS[pt * 3 + 0] = g0; gradS[pt * 3 + 1] = g1; gradS[pt * 3 + 2] = g2;
    out[5 * P + pi * 3 + 0] = g0;
    out[5 * P + pi * 3 + 1] = g1;
    out[5 * P + pi * 3 + 2] = g2;
    const float len = sqrtf(g0 * g0 + g1 * g1 + g2 * g2);
    const float il  = 1.f / fmaxf(len, 1e-12f);
    out[8 * P + pi * 3 + 0] = g0 * il;
    out[8 * P + pi * 3 + 1] = g1 * il;
    out[8 * P + pi * 3 + 2] = g2 * il;

    const float sdf = outt[pt * 16];
    const float be  = fabsf(beta[0]) + 1e-4f;
    const float sg  = (float)((sdf > 0.f) - (sdf < 0.f));
    out[3 * P + pi] = (0.5f + 0.5f * sg * expm1f(-fabsf(sdf) / be)) / be;
    out[4 * P + pi] = sdf;

    const float ddx = directions[pi * 3 + 0];
    const float ddy = directions[pi * 3 + 1];
    const float ddz = directions[pi * 3 + 2];
    const float tc  = ddx * g0 + ddy * g1 + ddz * g2;
    const float iter_cos = -fmaxf(-tc, 0.f);          // COS_ANNEAL == 1
    const float dl  = deltas[pi];
    const float en  = sdf + iter_cos * dl * 0.5f;
    const float ep  = sdf - iter_cos * dl * 0.5f;
    float inv_s = expf(variance[0] * 10.f);
    inv_s = fminf(fmaxf(inv_s, 1e-6f), 1e6f);
    const float pc = 1.f / (1.f + expf(-ep * inv_s));
    const float nc = 1.f / (1.f + expf(-en * inv_s));
    out[11 * P + pi] = fminf(fmaxf((pc - nc + 1e-5f) / (pc + 1e-5f), 0.f), 1.f);
  }
  __syncthreads();

  // ---- RGB input X2 = [p, grad, feats, geo, dir, pad] ----
  for (int i = lane; i < 16 * 64; i += 32) {
    const int pt = i >> 6, c = i & 63;
    float v = 0.f;
    if (c < 3)       v = points[(p0 + pt) * 3 + c];
    else if (c < 6)  v = gradS[pt * 3 + (c - 3)];
    else if (c < 38) v = featsS[pt * 32 + (c - 6)];
    else if (c < 53) v = outt[pt * 16 + (c - 37)];      // geo = cols 1..15
    else if (c < 56) v = directions[(p0 + pt) * 3 + (c - 53)];
    Xb[i] = f2bf(v);
  }
  __syncthreads();

  // ---- rgb layer 1: X2 @ Wr0 -> hr(16x256), relu ----
  {
    const v16bf a0 = load_a(Xb, 64, 0, lane);
    const v16bf a1 = load_a(Xb, 64, 32, lane);
#pragma unroll
    for (int nt = 0; nt < 16; ++nt) {
      v8f acc = {0.f, 0.f, 0.f, 0.f, 0.f, 0.f, 0.f, 0.f};
      acc = wmma_bf16(a0, load_b(ws + OFF_WTR0, 64, nt * 16, 0, lane), acc);
      acc = wmma_bf16(a1, load_b(ws + OFF_WTR0, 64, nt * 16, 32, lane), acc);
      store_tile(actA, 256, nt * 16, acc, br0, true, lane);
    }
  }
  __syncthreads();

  // ---- rgb layer 2: hr @ Wr1 -> sigmoid -> rgb ----
  {
    v16bf h[8];
#pragma unroll
    for (int k = 0; k < 8; ++k) h[k] = load_a(actA, 256, k * 32, lane);
    v8f acc = {0.f, 0.f, 0.f, 0.f, 0.f, 0.f, 0.f, 0.f};
#pragma unroll
    for (int k = 0; k < 8; ++k)
      acc = wmma_bf16(h[k], load_b(ws + OFF_WTR1, 256, 0, k * 32, lane), acc);
    const int n = lane & 15, mo = (lane >> 4) * 8;
    if (n < 3) {
      const float bb = br1[n];
#pragma unroll
      for (int r = 0; r < 8; ++r) {
        const float s = 1.f / (1.f + expf(-(acc[r] + bb)));
        out[(size_t)(p0 + mo + r) * 3 + n] = s;
      }
    }
  }
}

// ---------------- host entry ----------------
extern "C" void kernel_launch(void* const* d_in, const int* in_sizes, int n_in,
                              void* d_out, int out_size, void* d_ws,
                              size_t ws_size, hipStream_t stream) {
  (void)in_sizes; (void)n_in; (void)out_size; (void)ws_size;
  const float* points     = (const float*)d_in[0];
  const float* directions = (const float*)d_in[1];
  const float* deltas     = (const float*)d_in[2];
  const float* vol0       = (const float*)d_in[3];
  const float* vol1       = (const float*)d_in[4];
  const float* W0         = (const float*)d_in[5];
  const float* b0         = (const float*)d_in[6];
  const float* W1         = (const float*)d_in[7];
  const float* b1         = (const float*)d_in[8];
  const float* W2         = (const float*)d_in[9];
  const float* b2         = (const float*)d_in[10];
  const float* Wr0        = (const float*)d_in[11];
  const float* br0        = (const float*)d_in[12];
  const float* Wr1        = (const float*)d_in[13];
  const float* br1        = (const float*)d_in[14];
  const float* beta       = (const float*)d_in[15];
  const float* variance   = (const float*)d_in[16];
  unsigned short* ws = (unsigned short*)d_ws;
  float* out = (float*)d_out;

  prep_weights<<<(WS_TOTAL + 255) / 256, 256, 0, stream>>>(W0, W1, W2, Wr0,
                                                           Wr1, ws);
  sdf_field_kernel<<<NPTS / 16, 32, 0, stream>>>(
      points, directions, deltas, vol0, vol1, b0, b1, b2, br0, br1, W2, beta,
      variance, ws, out);
}
